// DeformAttentionBlock_87634512707709
// MI455X (gfx1250) — compile-verified, matrix operand order checked by reading, not run on hardware
//
#include <hip/hip_runtime.h>

#define BATCH 4
#define CH    128
#define HH    128
#define WW2   128
#define HW    (HH*WW2)      // 16384
#define K9    9
#define KC    (K9*CH)       // 1152
#define NPIX  (BATCH*HW)    // 65536
#define PIX_PER_WG 64

#define LDS_ATILE_BYTES (PIX_PER_WG*KC*2)            // 147456
#define LDS_OFFBUF_BYTES (PIX_PER_WG*18*4)           // 4608
#define LDS_TOTAL (LDS_ATILE_BYTES + LDS_OFFBUF_BYTES + PIX_PER_WG*4) // 152320

typedef __attribute__((ext_vector_type(8)))  float  v8f;
typedef __attribute__((ext_vector_type(16))) __bf16 v16bf;

union ABFrag { v16bf v; uint4 q[2]; };

__device__ __forceinline__ unsigned short f2bf(float f) {
  unsigned int u = __float_as_uint(f);
  u += 0x7FFFu + ((u >> 16) & 1u);           // round-to-nearest-even
  return (unsigned short)(u >> 16);
}

__device__ __forceinline__ void unpack8(uint4 q, float* f) {
  unsigned int u0=q.x,u1=q.y,u2=q.z,u3=q.w;
  f[0]=__uint_as_float(u0<<16); f[1]=__uint_as_float(u0&0xFFFF0000u);
  f[2]=__uint_as_float(u1<<16); f[3]=__uint_as_float(u1&0xFFFF0000u);
  f[4]=__uint_as_float(u2<<16); f[5]=__uint_as_float(u2&0xFFFF0000u);
  f[6]=__uint_as_float(u3<<16); f[7]=__uint_as_float(u3&0xFFFF0000u);
}

// ---------------- Prep 1: NCHW f32 -> NHWC bf16 (channel-contiguous for gathers)
__global__ void nchw_to_nhwc_bf16(const float* __restrict__ x, unsigned short* __restrict__ xn) {
  __shared__ float tile[32][33];
  int b  = blockIdx.z;
  int c0 = blockIdx.y * 32;
  int p0 = blockIdx.x * 32;
  int tx = threadIdx.x, ty = threadIdx.y;
  for (int i = ty; i < 32; i += 8)
    tile[i][tx] = x[((size_t)(b*CH + c0 + i))*HW + p0 + tx];
  __syncthreads();
  for (int i = ty; i < 32; i += 8)
    xn[((size_t)(b*HW + p0 + i))*CH + c0 + tx] = f2bf(tile[tx][i]);
}

// ---------------- Prep 2: repack weights to Bt[o][k*128+c] bf16 (WMMA-B friendly)
__global__ void build_bt(const float* __restrict__ w_def, const float* __restrict__ w_off,
                         unsigned short* __restrict__ bt_def, unsigned short* __restrict__ bt_off) {
  int idx = blockIdx.x*256 + threadIdx.x;
  const int ND = CH*KC;
  if (idx < ND) {
    int o = idx / KC, t = idx % KC, k = t / CH, c = t % CH;
    bt_def[idx] = f2bf(w_def[(o*CH + c)*K9 + k]);
  } else if (idx < ND + 32*KC) {
    int j = idx - ND;
    int o = j / KC, t = j % KC, k = t / CH, c = t % CH;
    bt_off[j] = (o < 18) ? f2bf(w_off[(o*CH + c)*K9 + k]) : (unsigned short)0;
  }
}

// ---------------- Fused: offset GEMM -> deformable gather -> main GEMM -> attn gate
__global__ __launch_bounds__(256)
void deform_attn_fused(const unsigned short* __restrict__ xn,
                       const unsigned short* __restrict__ bt_off,
                       const unsigned short* __restrict__ bt_def,
                       const float* __restrict__ b_off,
                       const float* __restrict__ b_def,
                       const float* __restrict__ w_attn,
                       const float* __restrict__ b_attn,
                       float* __restrict__ out) {
  extern __shared__ char smem[];
  unsigned short* Atile = (unsigned short*)smem;                        // [64][1152] bf16
  float* offbuf = (float*)(smem + LDS_ATILE_BYTES);                     // [64][18]
  float* attnb  = (float*)(smem + LDS_ATILE_BYTES + LDS_OFFBUF_BYTES);  // [64]

  int tid  = threadIdx.x;
  int pix0 = blockIdx.x * PIX_PER_WG;
  if (tid < PIX_PER_WG) attnb[tid] = 0.f;

  int wave = tid >> 5, lane = tid & 31;
  int m    = lane & 15;
  int kb   = (lane < 16) ? 0 : 8;           // 16-bit WMMA K sub-block by lane half
  int moff = (lane < 16) ? 0 : 8;

  // ---- Phase 0: regular 3x3 im2col into Atile via CDNA5 async global->LDS copies
  for (int t = tid; t < PIX_PER_WG*K9*16; t += 256) {
    int px  = t / (K9*16);
    int rem = t % (K9*16);
    int k   = rem / 16;
    int c0  = (rem & 15) * 8;
    int p  = pix0 + px;
    int b  = p >> 14;
    int hw = p & (HW-1);
    int h = hw >> 7, w = hw & 127;
    int hh = h - 1 + k/3, ww = w - 1 + k%3;
    unsigned short* dst = Atile + px*KC + k*CH + c0;
    if ((unsigned)hh < (unsigned)HH && (unsigned)ww < (unsigned)WW2) {
      const unsigned short* src = xn + ((size_t)(b*HW + hh*WW2 + ww))*CH + c0;
      unsigned ldsb = (unsigned)(unsigned long long)dst;   // LDS offset = low 32 bits of flat addr
      asm volatile("global_load_async_to_lds_b128 %0, %1, off"
                   :: "v"(ldsb), "v"(src) : "memory");
    } else {
      *(uint4*)dst = make_uint4(0u,0u,0u,0u);
    }
  }
  asm volatile("s_wait_asynccnt 0x0" ::: "memory");
  __syncthreads();

  // ---- Phase 1: offset conv GEMM (M=64 px, N=32 [18 used], K=1152), 1 tile/wave
  {
    int mt = wave >> 1, nt = wave & 1;
    v8f acc = {};
    const unsigned short* brow  = bt_off + (size_t)(nt*16 + m)*KC;
    const unsigned short* arow0 = Atile + (mt*16 + m)*KC;
    for (int s = 0; s < KC/32; ++s) {
      int kk = s*32;
      ABFrag a, bf;
      bf.q[0] = *(const uint4*)(brow + kk + kb);
      bf.q[1] = *(const uint4*)(brow + kk + kb + 16);
      a.q[0]  = *(const uint4*)(arow0 + kk + kb);
      a.q[1]  = *(const uint4*)(arow0 + kk + kb + 16);
      acc = __builtin_amdgcn_wmma_f32_16x16x32_bf16(false, a.v, false, bf.v, (short)0, acc, false, false);
    }
    int o = nt*16 + m;
    if (o < 18) {
      float bias = b_off[o];
      for (int r = 0; r < 8; ++r)
        offbuf[(mt*16 + r + moff)*18 + o] = acc[r] + bias;
    }
  }
  __syncthreads();   // offsets ready; Atile reads done -> safe to overwrite

  // ---- Phase 2: deformable bilinear gather, bf16-pack into im2col rows (overwrite Atile)
  for (int t = tid; t < PIX_PER_WG*K9*16; t += 256) {
    int px  = t / (K9*16);
    int rem = t % (K9*16);
    int k   = rem / 16;
    int c0  = (rem & 15) * 8;
    int p  = pix0 + px;
    int b  = p >> 14;
    int hw = p & (HW-1);
    int h = hw >> 7, w = hw & 127;
    float dy = offbuf[px*18 + 2*k];
    float dx = offbuf[px*18 + 2*k + 1];
    float ys = (float)(h - 1 + k/3) + dy;
    float xs = (float)(w - 1 + k%3) + dx;
    float y0f = floorf(ys), x0f = floorf(xs);
    float wy = ys - y0f, wx = xs - x0f;
    int y0 = (int)y0f, x0 = (int)x0f;
    const unsigned short* xb = xn + ((size_t)b * HW) * CH;
    float v00[8], v01[8], v10[8], v11[8];
#define LOAD8(dst, yy, xx) { uint4 q = make_uint4(0u,0u,0u,0u); \
      if ((unsigned)(yy) < (unsigned)HH && (unsigned)(xx) < (unsigned)WW2) \
        q = *(const uint4*)(xb + ((size_t)((yy)*WW2 + (xx)))*CH + c0); \
      unpack8(q, dst); }
    LOAD8(v00, y0,   x0  )
    LOAD8(v01, y0,   x0+1)
    LOAD8(v10, y0+1, x0  )
    LOAD8(v11, y0+1, x0+1)
#undef LOAD8
    float iwy = 1.f - wy, iwx = 1.f - wx;
    unsigned int qo[4];
    for (int j = 0; j < 4; ++j) {
      float r0 = (v00[2*j  ]*iwx + v01[2*j  ]*wx)*iwy + (v10[2*j  ]*iwx + v11[2*j  ]*wx)*wy;
      float r1 = (v00[2*j+1]*iwx + v01[2*j+1]*wx)*iwy + (v10[2*j+1]*iwx + v11[2*j+1]*wx)*wy;
      qo[j] = (unsigned int)f2bf(r0) | ((unsigned int)f2bf(r1) << 16);
    }
    *(uint4*)(Atile + px*KC + k*CH + c0) = make_uint4(qo[0],qo[1],qo[2],qo[3]);
  }
  __syncthreads();

  // ---- Phase 3: main GEMM. wave -> 16-wide N tile, 4 M tiles of 16 pixels
  int n0 = wave * 16;
  int o  = n0 + m;
  float bias = b_def[o];
  v8f acc[4];
  for (int mt = 0; mt < 4; ++mt)
    for (int j = 0; j < 8; ++j) acc[mt][j] = bias;
  const unsigned short* brow = bt_def + (size_t)o * KC;
  for (int s = 0; s < KC/32; ++s) {
    int kk = s*32;
    ABFrag bf;
    bf.q[0] = *(const uint4*)(brow + kk + kb);
    bf.q[1] = *(const uint4*)(brow + kk + kb + 16);
    for (int mt = 0; mt < 4; ++mt) {
      ABFrag a;
      const unsigned short* arow = Atile + (mt*16 + m)*KC + kk;
      a.q[0] = *(const uint4*)(arow + kb);
      a.q[1] = *(const uint4*)(arow + kb + 16);
      acc[mt] = __builtin_amdgcn_wmma_f32_16x16x32_bf16(false, a.v, false, bf.v, (short)0, acc[mt], false, false);
    }
  }

  // ---- Phase 4: attention partials (reduce over this wave's 16 output channels)
  float wa = w_attn[o];
  for (int mt = 0; mt < 4; ++mt)
    for (int r = 0; r < 8; ++r) {
      float v = acc[mt][r] * wa;
      v += __shfl_xor(v, 1);
      v += __shfl_xor(v, 2);
      v += __shfl_xor(v, 4);
      v += __shfl_xor(v, 8);
      if (m == 0) atomicAdd(&attnb[mt*16 + r + moff], v);
    }
  __syncthreads();

  // ---- Phase 5: sigmoid gate + relu + coalesced store (NCHW out)
  float battn = b_attn[0];
  int b = pix0 >> 14;
  for (int mt = 0; mt < 4; ++mt) {
    int hw0 = (pix0 & (HW-1)) + mt*16 + moff;
    float* dst = out + (((size_t)(b*CH + o)) << 14) + hw0;
    float vals[8];
    for (int r = 0; r < 8; ++r) {
      float a  = attnb[mt*16 + r + moff] + battn;
      float sg = 1.f / (1.f + __expf(-a));
      float vv = acc[mt][r] * sg;
      vals[r] = vv > 0.f ? vv : 0.f;
    }
    *(float4*)(dst)     = make_float4(vals[0],vals[1],vals[2],vals[3]);
    *(float4*)(dst + 4) = make_float4(vals[4],vals[5],vals[6],vals[7]);
  }
}

extern "C" void kernel_launch(void* const* d_in, const int* in_sizes, int n_in,
                              void* d_out, int out_size, void* d_ws, size_t ws_size,
                              hipStream_t stream) {
  const float* x      = (const float*)d_in[0];
  const float* w_off  = (const float*)d_in[1];
  const float* b_off  = (const float*)d_in[2];
  const float* w_def  = (const float*)d_in[3];
  const float* b_def  = (const float*)d_in[4];
  const float* w_attn = (const float*)d_in[5];
  const float* b_attn = (const float*)d_in[6];
  float* out = (float*)d_out;

  char* ws = (char*)d_ws;
  unsigned short* xn    = (unsigned short*)ws;                    // 16,777,216 B
  unsigned short* btdef = (unsigned short*)(ws + 16777216);       //    294,912 B
  unsigned short* btoff = (unsigned short*)(ws + 16777216 + 294912); // 73,728 B

  dim3 tb(32, 8);
  dim3 tg(HW/32, CH/32, BATCH);
  nchw_to_nhwc_bf16<<<tg, tb, 0, stream>>>(x, xn);

  int nbt = (CH + 32) * KC;
  build_bt<<<(nbt + 255)/256, 256, 0, stream>>>(w_def, w_off, btdef, btoff);

  deform_attn_fused<<<NPIX/PIX_PER_WG, 256, LDS_TOTAL, stream>>>(
      xn, btoff, btdef, b_off, b_def, w_attn, b_attn, out);
}
